// SelfAttention_49143015800838
// MI455X (gfx1250) — matrix operand order repaired and checked by reading, not compiled, not run on hardware
//
#include <hip/hip_runtime.h>
#include <hip/hip_bf16.h>

// ---------------------------------------------------------------------------
// Self-attention forward for MI455X (gfx1250)
// wave32, v_wmma_f32_16x16x32_f16, TDM tensor_load_to_lds double buffering
// B=2, T=2048, EMB=1024, HEADS=16, DH=64
// ---------------------------------------------------------------------------

typedef _Float16 half8  __attribute__((ext_vector_type(8)));
typedef _Float16 half16 __attribute__((ext_vector_type(16)));
typedef float    float8 __attribute__((ext_vector_type(8)));

#define BB    2
#define TT    2048
#define EMB   1024
#define HEADS 16
#define DH    64
#define MTOT  (BB * TT)       // 4096
#define NQKV  (3 * EMB)       // 3072

#if defined(__has_builtin)
#if __has_builtin(__builtin_amdgcn_tensor_load_to_lds)
#define HAVE_TDM 1
#endif
#endif
#ifndef HAVE_TDM
#define HAVE_TDM 0
#endif

__device__ __forceinline__ half16 cat8(half8 lo, half8 hi) {
  return __builtin_shufflevector(lo, hi, 0, 1, 2, 3, 4, 5, 6, 7,
                                         8, 9, 10, 11, 12, 13, 14, 15);
}

__device__ __forceinline__ float8 wmma_f16(half16 a, half16 b, float8 c) {
  return __builtin_amdgcn_wmma_f32_16x16x32_f16(false, a, false, b, (short)0, c,
                                                false, false);
}

#if HAVE_TDM
typedef unsigned int u32x4 __attribute__((ext_vector_type(4)));
typedef int          i32x8 __attribute__((ext_vector_type(8)));
typedef int          i32x4 __attribute__((ext_vector_type(4)));

typedef __attribute__((address_space(3))) const void lds_cv;

__device__ __forceinline__ unsigned lds_off(const void* p) {
  return (unsigned)(unsigned long long)(lds_cv*)p;
}

// 2D tile DMA: tile_w x tile_h elements of 2 bytes, row stride `stride`
// (elements), from global tile-start `gptr` into LDS byte offset `ldsb`.
// D# layout per CDNA5 ISA 8.3/8.4 (group0: count/lds/global/type,
// group1: data_size | dims | tile dims | stride).  This toolchain exposes the
// 6-argument builtin: (g0 u32x4, g1 i32x8, g2 i32x4, g3 i32x4, i32x8, cpol).
__device__ __forceinline__ void tdm_load_2d(const void* gptr, unsigned ldsb,
                                            unsigned tensor_w, unsigned tensor_h,
                                            unsigned tile_w, unsigned tile_h,
                                            unsigned stride) {
  unsigned long long ga = (unsigned long long)gptr;
  u32x4 g0 = {
      1u,                                               // count=1 (user D#)
      ldsb,                                             // lds_addr (bytes)
      (unsigned)(ga & 0xFFFFFFFFu),                     // global_addr[31:0]
      (unsigned)((ga >> 32) & 0x1FFFFFFu) | (2u << 30)  // addr[56:32] | type=2
  };
  i32x8 g1 = {
      (int)(1u << 16),                                   // data_size = 2 bytes
      (int)((tensor_w & 0xFFFFu) << 16),                 // tensor_dim0 lo16
      (int)(((tensor_w >> 16) & 0xFFFFu) |
            ((tensor_h & 0xFFFFu) << 16)),               // dim0 hi | dim1 lo
      (int)(((tensor_h >> 16) & 0xFFFFu) |
            ((tile_w & 0xFFFFu) << 16)),                 // dim1 hi | tile_dim0
      (int)(tile_h & 0xFFFFu),                           // tile_dim1 (dim2=0)
      (int)stride,                                       // dim0_stride lo32
      0, 0};
  i32x4 gz4 = {0, 0, 0, 0};
  i32x8 gz8 = {0, 0, 0, 0, 0, 0, 0, 0};
  __builtin_amdgcn_tensor_load_to_lds(g0, g1, gz4, gz4, gz8, 0);
}
#endif

// --------------------------- f32 -> f16 convert ----------------------------
__global__ void cvt_f32_to_f16(const float* __restrict__ in,
                               _Float16* __restrict__ out, int n) {
  int i = blockIdx.x * blockDim.x + threadIdx.x;
  int stride = gridDim.x * blockDim.x;
  for (; i < n; i += stride) out[i] = (_Float16)in[i];
}

// ----------------------- shared 128x128 GEMM core --------------------------
// acc[2][4] += X[128x1024 tile rows m0..] * W[n0..n0+127, :]^T, f16 WMMA.
__device__ __forceinline__ void gemm_core(const _Float16* __restrict__ X,
                                          const _Float16* __restrict__ W,
                                          int m0, int n0, float8 (&acc)[2][4],
                                          _Float16 (&As)[2][128][32],
                                          _Float16 (&Bs)[2][128][32]) {
  const int tid = threadIdx.x;
  const int lane = tid & 31;
  const int wave = tid >> 5;          // 0..7
  const int wm = (wave >> 1) * 32;    // 0,32,64,96
  const int wn = (wave & 1) * 64;     // 0,64
  const int arow = lane & 15;
  const int kb8 = (lane >> 4) * 8;
  const int bcol = lane & 15;
  const int kb16 = (lane >> 4) * 16;
  constexpr int NK = EMB / 32;

#if HAVE_TDM
  if (wave == 0) {
    tdm_load_2d(X + (size_t)m0 * EMB, lds_off(&As[0][0][0]), EMB, 128, 32, 128,
                EMB);
    tdm_load_2d(W + (size_t)n0 * EMB, lds_off(&Bs[0][0][0]), EMB, 128, 32, 128,
                EMB);
  }
  for (int t = 0; t < NK; ++t) {
    const int buf = t & 1;
    if (wave == 0) __builtin_amdgcn_s_wait_tensorcnt((short)0);
    __syncthreads();  // tile t resident; tile t-1 compute finished everywhere
    if (wave == 0 && t + 1 < NK) {
      tdm_load_2d(X + (size_t)m0 * EMB + (t + 1) * 32,
                  lds_off(&As[buf ^ 1][0][0]), EMB, 128, 32, 128, EMB);
      tdm_load_2d(W + (size_t)n0 * EMB + (t + 1) * 32,
                  lds_off(&Bs[buf ^ 1][0][0]), EMB, 128, 32, 128, EMB);
    }
    half16 afrag[2];
#pragma unroll
    for (int mb = 0; mb < 2; ++mb) {
      half8 lo = *(const half8*)&As[buf][wm + mb * 16 + arow][kb16];
      half8 hi = *(const half8*)&As[buf][wm + mb * 16 + arow][kb16 + 8];
      afrag[mb] = cat8(lo, hi);
    }
#pragma unroll
    for (int nb = 0; nb < 4; ++nb) {
      half8 blo = *(const half8*)&Bs[buf][wn + nb * 16 + bcol][kb16];
      half8 bhi = *(const half8*)&Bs[buf][wn + nb * 16 + bcol][kb16 + 8];
      half16 bfrag = cat8(blo, bhi);
#pragma unroll
      for (int mb = 0; mb < 2; ++mb)
        acc[mb][nb] = wmma_f16(afrag[mb], bfrag, acc[mb][nb]);
    }
  }
#else
  const int ldrow = tid >> 1;
  const int ldseg = (tid & 1) * 16;
  for (int k0 = 0; k0 < EMB; k0 += 32) {
    // issue all four b128 loads into distinct registers, then store to LDS
    const half8* sa = (const half8*)(X + (size_t)(m0 + ldrow) * EMB + k0 + ldseg);
    const half8* sb = (const half8*)(W + (size_t)(n0 + ldrow) * EMB + k0 + ldseg);
    half8 ra0 = sa[0], ra1 = sa[1], rb0 = sb[0], rb1 = sb[1];
    __syncthreads();  // previous tile fully consumed
    *(half8*)&As[0][ldrow][ldseg] = ra0;
    *(half8*)&As[0][ldrow][ldseg + 8] = ra1;
    *(half8*)&Bs[0][ldrow][ldseg] = rb0;
    *(half8*)&Bs[0][ldrow][ldseg + 8] = rb1;
    __syncthreads();
    half16 afrag[2];
#pragma unroll
    for (int mb = 0; mb < 2; ++mb) {
      half8 lo = *(const half8*)&As[0][wm + mb * 16 + arow][kb16];
      half8 hi = *(const half8*)&As[0][wm + mb * 16 + arow][kb16 + 8];
      afrag[mb] = cat8(lo, hi);
    }
#pragma unroll
    for (int nb = 0; nb < 4; ++nb) {
      half8 blo = *(const half8*)&Bs[0][wn + nb * 16 + bcol][kb16];
      half8 bhi = *(const half8*)&Bs[0][wn + nb * 16 + bcol][kb16 + 8];
      half16 bfrag = cat8(blo, bhi);
#pragma unroll
      for (int mb = 0; mb < 2; ++mb)
        acc[mb][nb] = wmma_f16(afrag[mb], bfrag, acc[mb][nb]);
    }
  }
#endif
}

// --------------------------- QKV projection GEMM ---------------------------
// Q,K scattered to [B,H,T,DH]; V scattered TRANSPOSED to [B,H,DH,T] so the
// attention P@V B-fragments become contiguous b128 loads.
__global__ __launch_bounds__(256) void gemm_qkv(
    const _Float16* __restrict__ X, const _Float16* __restrict__ W,
    _Float16* __restrict__ Qo, _Float16* __restrict__ Ko,
    _Float16* __restrict__ Vt) {
  __shared__ __align__(16) _Float16 As[2][128][32];
  __shared__ __align__(16) _Float16 Bs[2][128][32];

  const int m0 = blockIdx.y * 128;
  const int n0 = blockIdx.x * 128;
  const int lane = threadIdx.x & 31;
  const int wave = threadIdx.x >> 5;
  const int wm = (wave >> 1) * 32;
  const int wn = (wave & 1) * 64;
  const int bcol = lane & 15;
  const int rofs = 8 * (lane >> 4);

  float8 acc[2][4];
#pragma unroll
  for (int i = 0; i < 2; ++i)
#pragma unroll
    for (int j = 0; j < 4; ++j) acc[i][j] = (float8){};

  gemm_core(X, W, m0, n0, acc, As, Bs);

#pragma unroll
  for (int mb = 0; mb < 2; ++mb) {
#pragma unroll
    for (int g = 0; g < 8; ++g) {
      const int m = m0 + wm + mb * 16 + g + rofs;
      const int b = m >> 11;
      const int t = m & (TT - 1);
#pragma unroll
      for (int nb = 0; nb < 4; ++nb) {
        const int n = n0 + wn + nb * 16 + bcol;
        const int sect = n >> 10;
        const int h = (n & (EMB - 1)) >> 6;
        const int d = n & (DH - 1);
        const _Float16 val = (_Float16)acc[mb][nb][g];
        if (sect == 0)
          Qo[((size_t)(b * HEADS + h) * TT + t) * DH + d] = val;
        else if (sect == 1)
          Ko[((size_t)(b * HEADS + h) * TT + t) * DH + d] = val;
        else
          Vt[((size_t)(b * HEADS + h) * DH + d) * TT + t] = val;  // transposed
      }
    }
  }
}

// ----------------------------- flash attention -----------------------------
// grid: (T/64, B*HEADS); block: 128 threads = 4 waves; wave owns 16 query rows
__global__ __launch_bounds__(128) void attn_flash(
    const _Float16* __restrict__ Q, const _Float16* __restrict__ K,
    const _Float16* __restrict__ V, _Float16* __restrict__ O) {
  const int bh = blockIdx.y;
  const int tid = threadIdx.x;
  const int lane = tid & 31;
  const int wave = tid >> 5;
  const int qbase = blockIdx.x * 64 + wave * 16;

  const _Float16* q = Q + (size_t)bh * TT * DH;
  const _Float16* k = K + (size_t)bh * TT * DH;
  const _Float16* vt = V + (size_t)bh * DH * TT;  // [DH][T]

  __shared__ __align__(16) _Float16 Pl[4][16][32];

  const int arow = lane & 15;
  const int kb8 = (lane >> 4) * 8;
  const int bcol = lane & 15;
  const int kb16 = (lane >> 4) * 16;
  const int rofs = 8 * (lane >> 4);

  // Q A-fragments with 1/sqrt(DH)=0.125 folded in (exact in f16)
  half16 aq[2];
#pragma unroll
  for (int c = 0; c < 2; ++c) {
    const _Float16* base = q + (size_t)(qbase + arow) * DH + c * 32;
    half8 lo = *(const half8*)(base + kb16);
    half8 hi = *(const half8*)(base + kb16 + 8);
    aq[c] = cat8(lo, hi);
#pragma unroll
    for (int i = 0; i < 16; ++i) aq[c][i] *= (_Float16)0.125f;
  }

  float8 Oacc[4];
#pragma unroll
  for (int nb = 0; nb < 4; ++nb) Oacc[nb] = (float8){};
  float mrow[8], lrow[8];
#pragma unroll
  for (int g = 0; g < 8; ++g) { mrow[g] = -3.0e38f; lrow[g] = 0.0f; }

  for (int k0 = 0; k0 < qbase + 16; k0 += 32) {
    // S = Q @ K^T (32 keys, K-dim 64)
    float8 S[2];
#pragma unroll
    for (int nt = 0; nt < 2; ++nt) {
      const _Float16* kbase = k + (size_t)(k0 + nt * 16 + bcol) * DH;
      half16 b0 = cat8(*(const half8*)(kbase + kb16),
                       *(const half8*)(kbase + kb16 + 8));
      half16 b1 = cat8(*(const half8*)(kbase + 32 + kb16),
                       *(const half8*)(kbase + 32 + kb16 + 8));
      float8 c = (float8){};
      c = wmma_f16(aq[0], b0, c);
      c = wmma_f16(aq[1], b1, c);
      S[nt] = c;
    }

    // online softmax (row stats uniform within 16-lane groups)
    float p0v[8], p1v[8];
#pragma unroll
    for (int g = 0; g < 8; ++g) {
      const int row = qbase + g + rofs;
      float s0 = S[0][g];
      float s1 = S[1][g];
      if (k0 + bcol > row) s0 = -3.0e38f;
      if (k0 + 16 + bcol > row) s1 = -3.0e38f;
      float mx = fmaxf(s0, s1);
#pragma unroll
      for (int off = 1; off < 16; off <<= 1)
        mx = fmaxf(mx, __shfl_xor(mx, off, 32));
      const float mnew = fmaxf(mrow[g], mx);
      const float alpha = __expf(mrow[g] - mnew);
      const float p0 = __expf(s0 - mnew);
      const float p1 = __expf(s1 - mnew);
      float rs = p0 + p1;
#pragma unroll
      for (int off = 1; off < 16; off <<= 1) rs += __shfl_xor(rs, off, 32);
      lrow[g] = lrow[g] * alpha + rs;
      mrow[g] = mnew;
#pragma unroll
      for (int nb = 0; nb < 4; ++nb) Oacc[nb][g] *= alpha;
      p0v[g] = p0;
      p1v[g] = p1;
    }

    // re-layout P: C-fragment -> LDS -> A-fragment (per-wave tile)
#pragma unroll
    for (int g = 0; g < 8; ++g) {
      Pl[wave][g + rofs][bcol] = (_Float16)p0v[g];
      Pl[wave][g + rofs][16 + bcol] = (_Float16)p1v[g];
    }
    asm volatile("s_wait_dscnt 0x0" ::: "memory");
    half16 ap = cat8(*(const half8*)&Pl[wave][arow][kb16],
                     *(const half8*)&Pl[wave][arow][kb16 + 8]);

    // O += P @ V via transposed V: contiguous b128 loads per fragment
#pragma unroll
    for (int nb = 0; nb < 4; ++nb) {
      const _Float16* vb = vt + (size_t)(nb * 16 + bcol) * TT + k0 + kb16;
      half16 bv = cat8(*(const half8*)vb, *(const half8*)(vb + 8));
      Oacc[nb] = wmma_f16(ap, bv, Oacc[nb]);
    }
  }

  // normalize and store as [B, T, H*DH] f16
  const int b = bh >> 4;
  const int h = bh & 15;
#pragma unroll
  for (int g = 0; g < 8; ++g) {
    const float inv = 1.0f / lrow[g];
    const int t = qbase + g + rofs;
#pragma unroll
    for (int nb = 0; nb < 4; ++nb) {
      const int d = nb * 16 + bcol;
      O[(size_t)(b * TT + t) * EMB + h * DH + d] = (_Float16)(Oacc[nb][g] * inv);
    }
  }
}

// --------------------------- output projection -----------------------------
__global__ __launch_bounds__(256) void gemm_out(
    const _Float16* __restrict__ A, const _Float16* __restrict__ W,
    float* __restrict__ Y) {
  __shared__ __align__(16) _Float16 As[2][128][32];
  __shared__ __align__(16) _Float16 Bs[2][128][32];

  const int m0 = blockIdx.y * 128;
  const int n0 = blockIdx.x * 128;
  const int lane = threadIdx.x & 31;
  const int wave = threadIdx.x >> 5;
  const int wm = (wave >> 1) * 32;
  const int wn = (wave & 1) * 64;
  const int bcol = lane & 15;
  const int rofs = 8 * (lane >> 4);

  float8 acc[2][4];
#pragma unroll
  for (int i = 0; i < 2; ++i)
#pragma unroll
    for (int j = 0; j < 4; ++j) acc[i][j] = (float8){};

  gemm_core(A, W, m0, n0, acc, As, Bs);

#pragma unroll
  for (int mb = 0; mb < 2; ++mb)
#pragma unroll
    for (int g = 0; g < 8; ++g) {
      const int m = m0 + wm + mb * 16 + g + rofs;
#pragma unroll
      for (int nb = 0; nb < 4; ++nb) {
        const int n = n0 + wn + nb * 16 + bcol;
        Y[(size_t)m * EMB + n] = acc[mb][nb][g];
      }
    }
}

// ------------------------------- launcher ----------------------------------
extern "C" void kernel_launch(void* const* d_in, const int* in_sizes, int n_in,
                              void* d_out, int out_size, void* d_ws,
                              size_t ws_size, hipStream_t stream) {
  const float* x = (const float*)d_in[0];
  const float* w_qkv = (const float*)d_in[1];
  const float* w_out = (const float*)d_in[2];
  float* out = (float*)d_out;

  _Float16* ws = (_Float16*)d_ws;
  size_t off = 0;
  _Float16* xh = ws + off;     off += (size_t)MTOT * EMB;
  _Float16* wqkvh = ws + off;  off += (size_t)NQKV * EMB;
  _Float16* wouth = ws + off;  off += (size_t)EMB * EMB;
  _Float16* qh = ws + off;     off += (size_t)BB * HEADS * TT * DH;
  _Float16* kh = ws + off;     off += (size_t)BB * HEADS * TT * DH;
  _Float16* vth = ws + off;    off += (size_t)BB * HEADS * TT * DH;
  _Float16* ah = ws + off;     off += (size_t)MTOT * EMB;

  cvt_f32_to_f16<<<2048, 256, 0, stream>>>(x, xh, MTOT * EMB);
  cvt_f32_to_f16<<<2048, 256, 0, stream>>>(w_qkv, wqkvh, NQKV * EMB);
  cvt_f32_to_f16<<<1024, 256, 0, stream>>>(w_out, wouth, EMB * EMB);

  gemm_qkv<<<dim3(NQKV / 128, MTOT / 128), 256, 0, stream>>>(xh, wqkvh, qh, kh,
                                                             vth);
  attn_flash<<<dim3(TT / 64, BB * HEADS), 128, 0, stream>>>(qh, kh, vth, ah);
  gemm_out<<<dim3(EMB / 128, MTOT / 128), 256, 0, stream>>>(ah, wouth, out);
}
